// DiscriminatorCapsBiTCN_43628277792965
// MI455X (gfx1250) — matrix operand order repaired and checked by reading, not compiled, not run on hardware
//
#include <hip/hip_runtime.h>
#include <cstdint>

// ---------------------------------------------------------------------------
// CDNA5 (gfx1250) implementation of DiscriminatorCapsBiTCN.
// Conv1d -> implicit GEMM on v_wmma_f32_16x16x32_bf16 (wave32, one wave/WG).
//   * Activations channel-innermost (B, L, C) bf16: B fragments = 2x b128/lane
//   * Weights pre-swizzled to the per-lane A-fragment layout (bf16): 2x b128
//   * BN folded to per-channel (scale, shift) float2
//   * Epilogue: one packed b128 bf16 store per lane
// ---------------------------------------------------------------------------

typedef __attribute__((ext_vector_type(16))) __bf16 v16bf;
typedef __attribute__((ext_vector_type(8)))  __bf16 v8bf;
typedef __attribute__((ext_vector_type(8)))  float  v8f;

#define L_SEQ 2048
#define BATCH 64
#define LP    1024   // PrimaryCaps output length
#define NTG   16     // time-groups for partial-mean reduction

// ===========================================================================
// x (B, 64, L) f32  ->  (B, L, 64) bf16, channel-innermost for WMMA B-frags.
// ===========================================================================
__global__ __launch_bounds__(256) void xpose_to_bf16(
    const float* __restrict__ x, __bf16* __restrict__ xo)
{
    const int t = blockIdx.y * 256 + threadIdx.x;
    const int b = blockIdx.x;
    __bf16 row[64];
#pragma unroll
    for (int c = 0; c < 64; ++c)
        row[c] = (__bf16)x[((size_t)b * 64 + c) * L_SEQ + t];
    v8bf* dst = (v8bf*)(xo + ((size_t)b * L_SEQ + t) * 64);
#pragma unroll
    for (int g = 0; g < 8; ++g) {
        v8bf v;
#pragma unroll
        for (int e = 0; e < 8; ++e) v[e] = row[g * 8 + e];
        dst[g] = v;
    }
}

// ===========================================================================
// Weight pre-swizzle: w (COUT, CIN, NTAP) f32 -> wp bf16 laid out exactly as
// the per-lane WMMA 16x32 A fragments:
//   wp[(((tile*NTAP + j)*KB + kb)*32 + lane)*16 + e]
//     = w[(tile*16 + (lane&15))*CIN + kb*32 + (lane>>4)*16 + e, j]
// Run once per layer; conv waves then load fragments with 2x b128, coalesced.
// ===========================================================================
__global__ __launch_bounds__(256) void swizzle_w(
    const float* __restrict__ w, __bf16* __restrict__ wp,
    int CIN, int NTAP, int total)
{
    const int idx = blockIdx.x * 256 + threadIdx.x;
    if (idx >= total) return;
    const int e    = idx & 15;
    const int lane = (idx >> 4) & 31;
    int rest = idx >> 9;
    const int KB = CIN >> 5;
    const int kb = rest % KB; rest /= KB;
    const int j  = rest % NTAP; rest /= NTAP;
    const int tile = rest;
    const int row = lane & 15, hi = lane >> 4;
    const int k = kb * 32 + hi * 16 + e;
    wp[idx] = (__bf16)w[((size_t)(tile * 16 + row) * CIN + k) * NTAP + j];
}

// ===========================================================================
// BN fold: per-channel scale/shift so epilogue is one FMA.
//   sc = gamma / sqrt(var+eps); sh = (bias - mean)*sc + beta
// ===========================================================================
__global__ __launch_bounds__(64) void bn_prep(
    const float* __restrict__ bias, const float* __restrict__ gamma,
    const float* __restrict__ beta, const float* __restrict__ mean,
    const float* __restrict__ var, float2* __restrict__ ss, int C)
{
    const int c = blockIdx.x * 64 + threadIdx.x;
    if (c < C) {
        const float sc = gamma[c] * rsqrtf(var[c] + 1e-5f);
        ss[c] = make_float2(sc, (bias[c] - mean[c]) * sc + beta[c]);
    }
}

// ===========================================================================
// TCN block: dilated k=3 conv (causal or anti-causal) + folded-BN + ReLU
//            + residual (identity or fused 1x1 conv) + ReLU.
// Tap offset for weight j: sign * (2-j) * dil  (sign=-1 fwd causal, +1 bwd).
// One wave computes a 16(co) x 16(t) tile; TS consecutive t-tiles per wave
// so the A-fragments (held in VGPRs) are reused TS times.
// ===========================================================================
template <int CIN, int COUT, bool HAS_RES, int TS>
__global__ __launch_bounds__(32) void tcn_block_wmma(
    const __bf16* __restrict__ xin,   // (B, L, CIN) bf16
    __bf16*       __restrict__ xout,  // (B, L, COUT) bf16
    const __bf16* __restrict__ wp,    // pre-swizzled A fragments (NTAP=3)
    const __bf16* __restrict__ rwp,   // pre-swizzled 1x1 fragments (NTAP=1) or null
    const float2* __restrict__ ss,    // (COUT) folded BN scale/shift
    const float*  __restrict__ rb,    // (COUT) or null
    int dil, int sign)
{
    constexpr int KB = CIN / 32;          // K-blocks of 32 channels
    const int lane  = threadIdx.x & 31;
    const int tile  = blockIdx.x;
    const int co0   = tile * 16;
    const int hi    = lane >> 4;
    const int tbase = blockIdx.y * (16 * TS);
    const int b     = blockIdx.z;

    // ---- A fragments: direct aligned vector loads from pre-swizzled wp ----
    v16bf Aw[3][KB];
#pragma unroll
    for (int j = 0; j < 3; ++j)
#pragma unroll
        for (int kb = 0; kb < KB; ++kb)
            Aw[j][kb] = *(const v16bf*)(
                wp + ((size_t)((tile * 3 + j) * KB + kb) * 32 + lane) * 16);
    v16bf Ar[HAS_RES ? KB : 1];
    if (HAS_RES) {
#pragma unroll
        for (int kb = 0; kb < KB; ++kb)
            Ar[kb] = *(const v16bf*)(
                rwp + ((size_t)(tile * KB + kb) * 32 + lane) * 16);
    }

    for (int tt = 0; tt < TS; ++tt) {
        const int t0   = tbase + tt * 16;
        const int tcol = t0 + (lane & 15);    // this lane's output time (B col N)
        v8f acc  = {};
        v8f racc = {};
#pragma unroll
        for (int j = 0; j < 3; ++j) {
            const int  ts  = tcol + sign * (2 - j) * dil;
            const bool inb = (ts >= 0) && (ts < L_SEQ);
            const __bf16* src = xin + ((size_t)b * L_SEQ + ts) * CIN + hi * 16;
#pragma unroll
            for (int kb = 0; kb < KB; ++kb) {
                // B fragment: per lane 16 channel-consecutive bf16 (32B, aligned)
                v16bf bf = {};
                if (inb) bf = *(const v16bf*)(src + kb * 32);
                acc = __builtin_amdgcn_wmma_f32_16x16x32_bf16(
                        false, Aw[j][kb], false, bf, (short)0, acc, false, false);
                if (HAS_RES && j == 2) {  // tap j==2 has offset 0 == 1x1 conv tap
                    racc = __builtin_amdgcn_wmma_f32_16x16x32_bf16(
                        false, Ar[kb], false, bf, (short)0, racc, false, false);
                }
            }
        }
        // ---- fused epilogue: folded BN + ReLU + residual + ReLU ----
        v8bf rv;
        if (!HAS_RES)
            rv = *(const v8bf*)(xin + ((size_t)b * L_SEQ + tcol) * CIN + co0 + hi * 8);
        const float2* ssp = ss + co0 + hi * 8;
        v8bf ov;
#pragma unroll
        for (int r = 0; r < 8; ++r) {
            const float2 p = ssp[r];
            float y = fmaxf(acc[r] * p.x + p.y, 0.0f);
            const float res = HAS_RES ? (racc[r] + rb[co0 + hi * 8 + r])
                                      : (float)rv[r];
            y = fmaxf(y + res, 0.0f);
            ov[r] = (__bf16)y;
        }
        *(v8bf*)(xout + ((size_t)b * L_SEQ + tcol) * COUT + co0 + hi * 8) = ov;
    }
}

// ===========================================================================
// PrimaryCaps: conv1d(256->256, k=3, stride 2, pad 1) on concat(f,b), then
// per-capsule squash (8 dims live in the 8 accumulator rows of one lane)
// and partial mean over time written deterministically to workspace.
// ===========================================================================
__global__ __launch_bounds__(32) void primary_caps_wmma(
    const __bf16* __restrict__ f,    // (B, L, 128) bf16
    const __bf16* __restrict__ bb,   // (B, L, 128) bf16
    const __bf16* __restrict__ wp,   // pre-swizzled A fragments (NTAP=3, KB=8)
    const float*  __restrict__ bias, // (256)
    float*        __restrict__ partial)
{
    constexpr int KB = 8, TS = 4;
    const int lane = threadIdx.x & 31;
    const int hi   = lane >> 4;
    const int tile = blockIdx.x;        // 16 tiles cover 256 out channels
    const int co0  = tile * 16;
    const int tg   = blockIdx.y;        // 0..NTG-1
    const int b    = blockIdx.z;
    const int tp_base = tg * (16 * TS);

    v8f acc[TS] = {};
    for (int kb = 0; kb < KB; ++kb) {
        const int cib = kb * 32 + hi * 16;           // fragment channel base
        const __bf16* srcbuf = (cib < 128) ? f : bb; // 16-chunk never straddles
        const int cio = (cib < 128) ? cib : (cib - 128);
        v16bf Aw[3];
#pragma unroll
        for (int j = 0; j < 3; ++j)
            Aw[j] = *(const v16bf*)(
                wp + ((size_t)((tile * 3 + j) * KB + kb) * 32 + lane) * 16);
#pragma unroll
        for (int j = 0; j < 3; ++j) {
#pragma unroll
            for (int tt = 0; tt < TS; ++tt) {
                const int  tp  = tp_base + tt * 16 + (lane & 15);
                const int  ts  = 2 * tp - 1 + j;           // stride 2, pad 1
                const bool inb = (ts >= 0) && (ts < L_SEQ);
                v16bf bf = {};
                if (inb)
                    bf = *(const v16bf*)(srcbuf + ((size_t)b * L_SEQ + ts) * 128 + cio);
                acc[tt] = __builtin_amdgcn_wmma_f32_16x16x32_bf16(
                    false, Aw[j], false, bf, (short)0, acc[tt], false, false);
            }
        }
    }

    // ---- squash per (capsule, time); rows r = dims d of capsule co0/8+hi ----
    float csum[8];
#pragma unroll
    for (int r = 0; r < 8; ++r) csum[r] = 0.0f;
#pragma unroll
    for (int tt = 0; tt < TS; ++tt) {
        float a8[8];
        float n2 = 0.0f;
#pragma unroll
        for (int r = 0; r < 8; ++r) {
            const float a = acc[tt][r] + bias[co0 + hi * 8 + r];
            a8[r] = a;
            n2 += a * a;
        }
        const float fac = (n2 / (1.0f + n2)) / sqrtf(n2 + 1e-8f);
#pragma unroll
        for (int r = 0; r < 8; ++r) csum[r] += a8[r] * fac;
    }
    // reduce across the 16 lanes (time steps) of each half-wave
#pragma unroll
    for (int r = 0; r < 8; ++r) {
        float v = csum[r];
        for (int m = 8; m >= 1; m >>= 1) v += __shfl_xor(v, m, 16);
        csum[r] = v;
    }
    if ((lane & 15) == 0) {
        const int cap = co0 / 8 + hi;
#pragma unroll
        for (int r = 0; r < 8; ++r)
            partial[(((size_t)b * 32 + cap) * 8 + r) * NTG + tg] =
                csum[r] * (1.0f / (float)LP);
    }
}

// ===========================================================================
// Dynamic routing (3 iters) + classifier head. One 256-thread block per b.
// ===========================================================================
__global__ __launch_bounds__(256) void routing_head(
    const float* __restrict__ partial, // (B,32,8,NTG)
    const float* __restrict__ capsW,   // (32,16,16,8)
    const float* __restrict__ fc1w,    // (128,256)
    const float* __restrict__ fc1b,    // (128)
    const float* __restrict__ fc2w,    // (2,128)
    const float* __restrict__ fc2b,    // (2)
    float*       __restrict__ out)     // (B,2)
{
    __shared__ float xm[32][8];
    __shared__ float uh[32][16][16];
    __shared__ float blog[32][16];
    __shared__ float cc[32][16];
    __shared__ float sv[16][16];
    __shared__ float vv[16][16];
    __shared__ float h1[128];
    const int tid = threadIdx.x;
    const int b   = blockIdx.x;

    { // xm = sum of deterministic partials (already scaled by 1/LP)
        const int n = tid >> 3, d = tid & 7;
        if (tid < 256) {
            float s = 0.0f;
            for (int g = 0; g < NTG; ++g)
                s += partial[(((size_t)b * 32 + n) * 8 + d) * NTG + g];
            xm[n][d] = s;
        }
    }
    __syncthreads();
    // u_hat[n][o][d] = sum_i W[n,o,d,i] * xm[n][i]
    for (int idx = tid; idx < 32 * 16 * 16; idx += 256) {
        const int n = idx >> 8, o = (idx >> 4) & 15, d = idx & 15;
        float s = 0.0f;
#pragma unroll
        for (int i = 0; i < 8; ++i)
            s += capsW[(((size_t)(n * 16 + o) * 16) + d) * 8 + i] * xm[n][i];
        uh[n][o][d] = s;
    }
    for (int idx = tid; idx < 512; idx += 256) blog[idx >> 4][idx & 15] = 0.0f;
    __syncthreads();

    for (int it = 0; it < 3; ++it) {
        if (tid < 32) {  // softmax over o per n
            float mx = -1e30f;
            for (int o = 0; o < 16; ++o) mx = fmaxf(mx, blog[tid][o]);
            float e[16], sum = 0.0f;
            for (int o = 0; o < 16; ++o) { e[o] = expf(blog[tid][o] - mx); sum += e[o]; }
            const float inv = 1.0f / sum;
            for (int o = 0; o < 16; ++o) cc[tid][o] = e[o] * inv;
        }
        __syncthreads();
        if (tid < 256) {  // s[o][d] = sum_n c[n][o]*u_hat[n][o][d]
            const int o = tid >> 4, d = tid & 15;
            float s = 0.0f;
            for (int n = 0; n < 32; ++n) s += cc[n][o] * uh[n][o][d];
            sv[o][d] = s;
        }
        __syncthreads();
        if (tid < 16) {   // v = squash(s) over d
            float n2 = 0.0f;
            for (int d = 0; d < 16; ++d) n2 += sv[tid][d] * sv[tid][d];
            const float fac = (n2 / (1.0f + n2)) / sqrtf(n2 + 1e-8f);
            for (int d = 0; d < 16; ++d) vv[tid][d] = sv[tid][d] * fac;
        }
        __syncthreads();
        for (int idx = tid; idx < 512; idx += 256) {  // blog += <u_hat, v>
            const int n = idx >> 4, o = idx & 15;
            float s = 0.0f;
            for (int d = 0; d < 16; ++d) s += uh[n][o][d] * vv[o][d];
            blog[n][o] += s;
        }
        __syncthreads();
    }
    if (tid < 128) {  // fc1 + relu (flat k = o*16 + d)
        float s = fc1b[tid];
        for (int k = 0; k < 256; ++k)
            s += fc1w[(size_t)tid * 256 + k] * vv[k >> 4][k & 15];
        h1[tid] = fmaxf(s, 0.0f);
    }
    __syncthreads();
    if (tid < 2) {    // fc2
        float s = fc2b[tid];
        for (int j = 0; j < 128; ++j) s += fc2w[(size_t)tid * 128 + j] * h1[j];
        out[(size_t)b * 2 + tid] = s;
    }
}

// ===========================================================================
// Launcher
// ===========================================================================
namespace {
struct BP { const float *w, *b, *g, *be, *m, *v, *rw, *rb; };

static int get_blocks(void* const* d_in, int start, BP* blk, bool sortedOrder) {
    for (int i = 0; i < 4; ++i) {
        const bool hasres = (i == 2);  // 64 -> 128 block
        if (!sortedOrder) {  // dict insertion order: w,b,gamma,beta,mean,var[,rw,rb]
            blk[i].w  = (const float*)d_in[start + 0];
            blk[i].b  = (const float*)d_in[start + 1];
            blk[i].g  = (const float*)d_in[start + 2];
            blk[i].be = (const float*)d_in[start + 3];
            blk[i].m  = (const float*)d_in[start + 4];
            blk[i].v  = (const float*)d_in[start + 5];
            if (hasres) { blk[i].rw = (const float*)d_in[start + 6];
                          blk[i].rb = (const float*)d_in[start + 7]; start += 8; }
            else        { blk[i].rw = nullptr; blk[i].rb = nullptr;  start += 6; }
        } else {             // jax sorted keys: b,beta,gamma,mean[,rb,rw],var,w
            blk[i].b  = (const float*)d_in[start + 0];
            blk[i].be = (const float*)d_in[start + 1];
            blk[i].g  = (const float*)d_in[start + 2];
            blk[i].m  = (const float*)d_in[start + 3];
            if (hasres) {
                blk[i].rb = (const float*)d_in[start + 4];
                blk[i].rw = (const float*)d_in[start + 5];
                blk[i].v  = (const float*)d_in[start + 6];
                blk[i].w  = (const float*)d_in[start + 7];
                start += 8;
            } else {
                blk[i].v  = (const float*)d_in[start + 4];
                blk[i].w  = (const float*)d_in[start + 5];
                blk[i].rw = nullptr; blk[i].rb = nullptr;
                start += 6;
            }
        }
    }
    return start;
}
} // namespace

extern "C" void kernel_launch(void* const* d_in, const int* in_sizes, int n_in,
                              void* d_out, int out_size, void* d_ws, size_t ws_size,
                              hipStream_t stream)
{
    const float *X, *pcw, *pcb, *capsW, *fc1w, *fc1b, *fc2w, *fc2b;
    BP F[4], Bk[4];

    if (in_sizes[0] == 64 * 64 * 2048) {
        // top-level dict insertion order: x, then params (tcn_f, tcn_b, pc_w, ...)
        X = (const float*)d_in[0];
        int idx = 1;
        idx   = get_blocks(d_in, idx, F,  false);
        idx   = get_blocks(d_in, idx, Bk, false);
        pcw   = (const float*)d_in[idx++]; pcb  = (const float*)d_in[idx++];
        capsW = (const float*)d_in[idx++];
        fc1w  = (const float*)d_in[idx++]; fc1b = (const float*)d_in[idx++];
        fc2w  = (const float*)d_in[idx++]; fc2b = (const float*)d_in[idx++];
    } else {
        // jax tree_leaves sorted-key order: params leaves first, x last
        int idx = 0;
        capsW = (const float*)d_in[idx++];
        fc1b  = (const float*)d_in[idx++]; fc1w = (const float*)d_in[idx++];
        fc2b  = (const float*)d_in[idx++]; fc2w = (const float*)d_in[idx++];
        pcb   = (const float*)d_in[idx++]; pcw  = (const float*)d_in[idx++];
        idx   = get_blocks(d_in, idx, Bk, true);
        idx   = get_blocks(d_in, idx, F,  true);
        X = (const float*)d_in[idx++];
    }

    // ---- workspace layout ----
    __bf16* X16 = (__bf16*)d_ws;                       // (B, L, 64)  bf16
    const size_t xsz = (size_t)BATCH * L_SEQ * 64;
    const size_t asz = (size_t)BATCH * L_SEQ * 128;    // (B, L, 128) bf16
    __bf16* F0 = X16 + xsz;
    __bf16* F1 = F0 + asz;
    __bf16* B0 = F1 + asz;
    __bf16* B1 = B0 + asz;
    float*  PART = (float*)(B1 + asz);                 // (B,32,8,NTG) f32
    const size_t psz = (size_t)BATCH * 32 * 8 * NTG;
    float2* SS = (float2*)(PART + psz);                // folded BN, 768 float2
    float2* ssF[4] = { SS, SS + 64, SS + 128, SS + 256 };
    float2* ssB[4] = { SS + 384, SS + 448, SS + 512, SS + 640 };
    __bf16* WP = (__bf16*)(SS + 768);                  // pre-swizzled weights
    size_t off = 0;
    auto wpal = [&](size_t n) { __bf16* p = WP + off; off += n; return p; };
    __bf16* wpF[4] = { wpal(64*64*3), wpal(64*64*3), wpal(128*64*3), wpal(128*128*3) };
    __bf16* rwpF   = wpal(128*64);
    __bf16* wpB[4] = { wpal(64*64*3), wpal(64*64*3), wpal(128*64*3), wpal(128*128*3) };
    __bf16* rwpB   = wpal(128*64);
    __bf16* wpPC   = wpal(256*256*3);

    // ---- prep: input transpose, weight swizzle, BN fold ----
    xpose_to_bf16<<<dim3(BATCH, L_SEQ / 256), dim3(256), 0, stream>>>(X, X16);

    const int cinsF[4] = { 64, 64, 64, 128 };
    const int coutF[4] = { 64, 64, 128, 128 };
    for (int i = 0; i < 4; ++i) {
        const int totF = coutF[i] * cinsF[i] * 3;
        swizzle_w<<<(totF + 255) / 256, 256, 0, stream>>>(F[i].w,  wpF[i], cinsF[i], 3, totF);
        swizzle_w<<<(totF + 255) / 256, 256, 0, stream>>>(Bk[i].w, wpB[i], cinsF[i], 3, totF);
        bn_prep<<<(coutF[i] + 63) / 64, 64, 0, stream>>>(
            F[i].b,  F[i].g,  F[i].be,  F[i].m,  F[i].v,  ssF[i], coutF[i]);
        bn_prep<<<(coutF[i] + 63) / 64, 64, 0, stream>>>(
            Bk[i].b, Bk[i].g, Bk[i].be, Bk[i].m, Bk[i].v, ssB[i], coutF[i]);
    }
    swizzle_w<<<(128*64 + 255) / 256, 256, 0, stream>>>(F[2].rw,  rwpF, 64, 1, 128*64);
    swizzle_w<<<(128*64 + 255) / 256, 256, 0, stream>>>(Bk[2].rw, rwpB, 64, 1, 128*64);
    swizzle_w<<<(256*256*3 + 255) / 256, 256, 0, stream>>>(pcw, wpPC, 256, 3, 256*256*3);

    const dim3 wv(32);
    // ---- forward TCN stack (causal, sign=-1): X16->F0->F1->F0->F1 ----
    tcn_block_wmma<64, 64, false, 8><<<dim3(4, 16, BATCH), wv, 0, stream>>>(
        X16, F0, wpF[0], nullptr, ssF[0], nullptr, 1, -1);
    tcn_block_wmma<64, 64, false, 8><<<dim3(4, 16, BATCH), wv, 0, stream>>>(
        F0, F1, wpF[1], nullptr, ssF[1], nullptr, 2, -1);
    tcn_block_wmma<64, 128, true, 8><<<dim3(8, 16, BATCH), wv, 0, stream>>>(
        F1, F0, wpF[2], rwpF, ssF[2], F[2].rb, 4, -1);
    tcn_block_wmma<128, 128, false, 8><<<dim3(8, 16, BATCH), wv, 0, stream>>>(
        F0, F1, wpF[3], nullptr, ssF[3], nullptr, 8, -1);
    // ---- backward TCN stack (anti-causal, sign=+1): X16->B0->B1->B0->B1 ----
    tcn_block_wmma<64, 64, false, 8><<<dim3(4, 16, BATCH), wv, 0, stream>>>(
        X16, B0, wpB[0], nullptr, ssB[0], nullptr, 1, +1);
    tcn_block_wmma<64, 64, false, 8><<<dim3(4, 16, BATCH), wv, 0, stream>>>(
        B0, B1, wpB[1], nullptr, ssB[1], nullptr, 2, +1);
    tcn_block_wmma<64, 128, true, 8><<<dim3(8, 16, BATCH), wv, 0, stream>>>(
        B1, B0, wpB[2], rwpB, ssB[2], Bk[2].rb, 4, +1);
    tcn_block_wmma<128, 128, false, 8><<<dim3(8, 16, BATCH), wv, 0, stream>>>(
        B0, B1, wpB[3], nullptr, ssB[3], nullptr, 8, +1);

    // ---- PrimaryCaps conv + squash + partial mean ----
    primary_caps_wmma<<<dim3(16, NTG, BATCH), wv, 0, stream>>>(F1, B1, wpPC, pcb, PART);

    // ---- routing + classifier head ----
    routing_head<<<dim3(BATCH), dim3(256), 0, stream>>>(
        PART, capsW, fc1w, fc1b, fc2w, fc2b, (float*)d_out);
}
